// MotionModel_16149077033004
// MI455X (gfx1250) — compile-verified, matrix operand order checked by reading, not run on hardware
//
#include <hip/hip_runtime.h>
#include <hip/hip_bf16.h>

// Problem constants (from reference)
#define NB   8
#define DCH  32
#define HH   256
#define WW   256
#define HWP  (HH * WW)      // 65536
#define HIDC 64
#define KKC  49
#define AAC  256

typedef __attribute__((ext_vector_type(16))) __bf16 bf16x16;
typedef __attribute__((ext_vector_type(8)))  float  f32x8;
typedef __attribute__((ext_vector_type(4)))  unsigned int u32x4;
typedef __attribute__((ext_vector_type(8)))  unsigned int u32x8;

#define WMMA_BF16(A, B, C) \
  __builtin_amdgcn_wmma_f32_16x16x32_bf16(false, (A), false, (B), (short)0, (C), false, false)

__device__ __forceinline__ __bf16 f2bf(float f) { return (__bf16)f; }

__device__ __forceinline__ unsigned short bf_bits(float f) {
  __bf16 b = (__bf16)f;
  unsigned short s;
  __builtin_memcpy(&s, &b, 2);
  return s;
}

// ---------------------------------------------------------------------------
// Tensor Data Mover: stage a linear blob (units4 DWORDs) from global into LDS.
// D# per CDNA5 ISA ch.8: group0 = {count/lds_addr/global_addr/type},
// group1 = {mask/data_size, tensor_dim0/1, tile_dim0/1/2, strides}.
// 1-row 2D tile: tile_dim0 = tensor_dim0 = units4 (4-byte units), tile_dim1=1.
// Issued by one wave; completion tracked with TENSORcnt.
// ---------------------------------------------------------------------------
__device__ __forceinline__ void tdm_load_to_lds(unsigned lds_off, const void* gsrc,
                                                unsigned units4) {
  unsigned long long ga = (unsigned long long)gsrc;
  u32x4 g0;
  g0[0] = 1u;                                            // count=1, user mode
  g0[1] = lds_off;                                       // lds_addr (bytes)
  g0[2] = (unsigned)(ga & 0xFFFFFFFFu);                  // global_addr[31:0]
  g0[3] = (unsigned)((ga >> 32) & 0x01FFFFFFu) | (2u << 30);  // addr[56:32] | type=2
  u32x8 g1;
  g1[0] = (2u << 16);                                    // workgroup_mask=0, data_size=4B
  g1[1] = (units4 & 0xFFFFu) << 16;                      // tensor_dim0[15:0] @ bits47:32? -> [31:16]
  g1[2] = ((units4 >> 16) & 0xFFFFu) | (1u << 16);       // tensor_dim0[31:16] | tensor_dim1 lo=1
  g1[3] = (units4 & 0xFFFFu) << 16;                      // tensor_dim1 hi=0 | tile_dim0
  g1[4] = 1u;                                            // tile_dim1=1, tile_dim2=0
  g1[5] = units4;                                        // tensor_dim0_stride[31:0]
  g1[6] = 0u;                                            // stride hi / tensor_dim1_stride lo
  g1[7] = 0u;
  asm volatile("tensor_load_to_lds %0, %1" :: "s"(g0), "s"(g1) : "memory");
}

// ---------------------------------------------------------------------------
// Repack OIHW f32 weights into a WMMA-B-operand-ordered bf16 blob:
//   blob[((kt*NT + nt)*32 + lane)*16 + j] = W[n = nt*16 + (lane&15),
//                                             k = kt*32 + (lane>>4)*16 + j]
//   with K index = tap*I + ic (channel fastest within a tap).
// ---------------------------------------------------------------------------
__global__ void repack_weights(const float* __restrict__ w, unsigned short* __restrict__ blob,
                               int O, int I, int KH, int KW) {
  int Ktot = I * KH * KW;
  int KT = (Ktot + 31) / 32;
  int NT = (O + 15) / 16;
  int total = KT * NT * 32 * 16;
  for (int t = blockIdx.x * blockDim.x + threadIdx.x; t < total; t += gridDim.x * blockDim.x) {
    int j    = t & 15;
    int lane = (t >> 4) & 31;
    int tile = t >> 9;
    int nt   = tile % NT;
    int kt   = tile / NT;
    int n = nt * 16 + (lane & 15);
    int k = kt * 32 + (lane >> 4) * 16 + j;
    float v = 0.f;
    if (n < O && k < Ktot) {
      int tap = k / I, ic = k % I;
      v = w[((n * I + ic) * KH + tap / KW) * KW + (tap % KW)];
    }
    blob[t] = bf_bits(v);
  }
}

// ---------------------------------------------------------------------------
// lang = log_softmax(relu(action @ w_a1 + b_a1) @ w_a2 + b_a2)   (8 x 49)
// ---------------------------------------------------------------------------
__global__ void lang_kernel(const float* __restrict__ action, const float* __restrict__ w_a1,
                            const float* __restrict__ b_a1, const float* __restrict__ w_a2,
                            const float* __restrict__ b_a2, float* __restrict__ lang) {
  __shared__ float hid[HIDC];
  __shared__ float z[KKC];
  int n = blockIdx.x;
  int t = threadIdx.x;  // 0..63
  float acc = b_a1[t];
  for (int a = 0; a < AAC; ++a) acc += action[n * AAC + a] * w_a1[a * HIDC + t];
  hid[t] = fmaxf(acc, 0.f);
  __syncthreads();
  if (t < KKC) {
    float zz = b_a2[t];
    for (int h = 0; h < HIDC; ++h) zz += hid[h] * w_a2[h * KKC + t];
    z[t] = zz;
  }
  __syncthreads();
  if (t < KKC) {
    float mx = -INFINITY;
    for (int k = 0; k < KKC; ++k) mx = fmaxf(mx, z[k]);
    float s = 0.f;
    for (int k = 0; k < KKC; ++k) s += __expf(z[k] - mx);
    lang[n * 64 + t] = z[t] - (mx + __logf(s));
  }
}

// ---------------------------------------------------------------------------
// 2x2 maxpool, NCHW f32 (8,32,512,512) -> NHWC f32 x[n][p][c]
// ---------------------------------------------------------------------------
__global__ void maxpool_kernel(const float* __restrict__ sem, float* __restrict__ x) {
  int total = NB * DCH * HWP;  // 16.7M
  for (int t = blockIdx.x * blockDim.x + threadIdx.x; t < total; t += gridDim.x * blockDim.x) {
    int p = t & (HWP - 1);
    int c = (t >> 16) & (DCH - 1);
    int n = t >> 21;
    int i = p >> 8, j = p & 255;
    const float* base = sem + (((long)(n * DCH + c) * 512 + i * 2) * 512 + j * 2);
    float m = fmaxf(fmaxf(base[0], base[1]), fmaxf(base[512], base[513]));
    x[(long)(n * HWP + p) * DCH + c] = m;
  }
}

// ---------------------------------------------------------------------------
// h0 = relu(conv1x1(x, w_in) + b_in) : NHWC f32 32ch -> NHWC bf16 64ch
// One wave = one 16-pixel M-tile, 4 WMMA (K=32, N=64).
// ---------------------------------------------------------------------------
__global__ void conv1x1_in_kernel(const float* __restrict__ x, const unsigned short* __restrict__ blobW,
                                  const float* __restrict__ bias, unsigned short* __restrict__ h0) {
  int wave = blockIdx.x * (blockDim.x >> 5) + (threadIdx.x >> 5);
  int lane = threadIdx.x & 31;
  int n  = wave >> 12;        // 4096 M-tiles per image
  int mt = wave & 4095;
  long pixBase = (long)n * HWP + mt * 16;
  int m = lane & 15, half = lane >> 5 ? 0 : (lane >> 4);
  half = lane >> 4;

  const float* ap = x + (pixBase + m) * DCH + half * 16;
  bf16x16 a;
#pragma unroll
  for (int j = 0; j < 16; ++j) a[j] = f2bf(ap[j]);

  const bf16x16* bl = (const bf16x16*)blobW;  // KT=1, NT=4
  __builtin_prefetch((const void*)blobW, 0, 3);
#pragma unroll
  for (int nt = 0; nt < 4; ++nt) {
    f32x8 acc = {};
    bf16x16 b = bl[nt * 32 + lane];
    acc = WMMA_BF16(a, b, acc);
    int ch = nt * 16 + (lane & 15);
    float bi = bias[ch];
#pragma unroll
    for (int r = 0; r < 8; ++r) {
      int mm = r + half * 8;
      h0[(pixBase + mm) * HIDC + ch] = bf_bits(fmaxf(acc[r] + bi, 0.f));
    }
  }
}

// ---------------------------------------------------------------------------
// hout = relu(conv3x3(hin, W, dilation=dil, pad=dil) + bias), 64->64 ch.
// Implicit GEMM: K = 9 taps * 64 ch = 18 k-tiles; 72 WMMA per wave.
// The full repacked weight blob (72 KB) is staged into LDS once per
// workgroup via the Tensor Data Mover; B operands then come from ds_load.
// ---------------------------------------------------------------------------
__global__ void conv3x3_kernel(const unsigned short* __restrict__ hin,
                               const unsigned short* __restrict__ blobW,
                               const float* __restrict__ bias,
                               unsigned short* __restrict__ hout, int dil) {
  __shared__ __align__(128) unsigned short wlds[18 * 4 * 32 * 16];  // 73728 B

  // Stage weights via TDM (wave 0 only; TDM ignores EXEC, issued per-wave).
  if ((threadIdx.x >> 5) == 0) {
    tdm_load_to_lds((unsigned)(unsigned long long)(void*)wlds, blobW,
                    (18 * 4 * 32 * 16 * 2) / 4);
    __builtin_amdgcn_s_wait_tensorcnt(0);
  }
  __syncthreads();

  int wave = blockIdx.x * (blockDim.x >> 5) + (threadIdx.x >> 5);
  int lane = threadIdx.x & 31;
  int n  = wave >> 12;
  int mt = wave & 4095;
  int i  = mt >> 4;           // pixel row
  int j0 = (mt & 15) << 4;    // pixel col base
  int m = lane & 15, half = lane >> 4;
  int jm = j0 + m;

  f32x8 acc[4] = {{}, {}, {}, {}};

  for (int ty = 0; ty < 3; ++ty) {
    int yi = i + (ty - 1) * dil;
    bool rowok = (yi >= 0) && (yi < HH);
    for (int tx = 0; tx < 3; ++tx) {
      int xj = jm + (tx - 1) * dil;
      bool ok = rowok && (xj >= 0) && (xj < WW);
      long spix = (long)n * HWP + (long)yi * WW + xj;
      const unsigned short* ap0 = hin + spix * HIDC;
      int tap = ty * 3 + tx;
#pragma unroll
      for (int kh = 0; kh < 2; ++kh) {  // channel-half: ktile = tap*2 + kh
        bf16x16 a = {};
        if (ok) a = *(const bf16x16*)(ap0 + kh * 32 + half * 16);
        int kt = tap * 2 + kh;
#pragma unroll
        for (int nt = 0; nt < 4; ++nt) {
          bf16x16 b = *(const bf16x16*)&wlds[((kt * 4 + nt) * 32 + lane) * 16];
          acc[nt] = WMMA_BF16(a, b, acc[nt]);
        }
      }
    }
  }

  long pixBase = (long)n * HWP + mt * 16;
#pragma unroll
  for (int nt = 0; nt < 4; ++nt) {
    int ch = nt * 16 + (lane & 15);
    float bi = bias[ch];
#pragma unroll
    for (int r = 0; r < 8; ++r) {
      int mm = r + half * 8;
      hout[(pixBase + mm) * HIDC + ch] = bf_bits(fmaxf(acc[nt][r] + bi, 0.f));
    }
  }
}

// ---------------------------------------------------------------------------
// Fused tail GEMMs:
//   x2 = x + conv1x1(h2, w_out) + b_out          (32 ch, f32)
//   logits = conv1x1(x2, w_k) + b_k + lang[n]    (49 ch)
//   src[n][k][p] = logits                        (planar for gather)
// x2 goes through LDS to convert C-layout -> A-layout (bf16).
// ---------------------------------------------------------------------------
__global__ void out_motion_kernel(const unsigned short* __restrict__ h2,
                                  const float* __restrict__ x,
                                  const unsigned short* __restrict__ blobWo,
                                  const float* __restrict__ b_out,
                                  const unsigned short* __restrict__ blobWk,
                                  const float* __restrict__ b_k,
                                  const float* __restrict__ lang,
                                  float* __restrict__ src) {
  __shared__ __align__(64) unsigned short x2s[8][16][DCH];
  int wid  = threadIdx.x >> 5;
  int wave = blockIdx.x * (blockDim.x >> 5) + wid;
  int lane = threadIdx.x & 31;
  int n  = wave >> 12;
  int mt = wave & 4095;
  long pixBase = (long)n * HWP + mt * 16;
  int m = lane & 15, half = lane >> 4;

  // GEMM1: [16 pix, 64] @ [64, 32]
  const bf16x16* blo = (const bf16x16*)blobWo;  // KT=2, NT=2
  f32x8 accO[2] = {{}, {}};
#pragma unroll
  for (int kt = 0; kt < 2; ++kt) {
    bf16x16 a = *(const bf16x16*)(h2 + (pixBase + m) * HIDC + kt * 32 + half * 16);
#pragma unroll
    for (int nt = 0; nt < 2; ++nt) {
      bf16x16 b = blo[(kt * 2 + nt) * 32 + lane];
      accO[nt] = WMMA_BF16(a, b, accO[nt]);
    }
  }
  // x2 = acc + b_out + x ; stash bf16 tile into LDS (C-layout -> memory layout)
#pragma unroll
  for (int nt = 0; nt < 2; ++nt) {
    int ch = nt * 16 + (lane & 15);
    float bi = b_out[ch];
#pragma unroll
    for (int r = 0; r < 8; ++r) {
      int mm = r + half * 8;
      float v = accO[nt][r] + bi + x[(pixBase + mm) * DCH + ch];
      x2s[wid][mm][ch] = bf_bits(v);
    }
  }
  __syncthreads();

  // GEMM2: [16 pix, 32] @ [32, 64(pad of 49)]
  bf16x16 a2 = *(const bf16x16*)&x2s[wid][m][half * 16];
  const bf16x16* blk = (const bf16x16*)blobWk;  // KT=1, NT=4
  f32x8 accK[4] = {{}, {}, {}, {}};
#pragma unroll
  for (int nt = 0; nt < 4; ++nt) {
    bf16x16 b = blk[nt * 32 + lane];
    accK[nt] = WMMA_BF16(a2, b, accK[nt]);
  }
#pragma unroll
  for (int nt = 0; nt < 4; ++nt) {
    int ch = nt * 16 + (lane & 15);
    if (ch < KKC) {
      float add = b_k[ch] + lang[n * 64 + ch];
#pragma unroll
      for (int r = 0; r < 8; ++r) {
        int mm = r + half * 8;
        src[(long)(n * KKC + ch) * HWP + mt * 16 + mm] = accK[nt][r] + add;
      }
    }
  }
}

// ---------------------------------------------------------------------------
// Per-pixel: src[k][p] += log_belief[p] - logsumexp_k(src[:, p])
// ---------------------------------------------------------------------------
__global__ void normalize_kernel(const float* __restrict__ log_belief, float* __restrict__ src) {
  int total = NB * HWP;
  for (int t = blockIdx.x * blockDim.x + threadIdx.x; t < total; t += gridDim.x * blockDim.x) {
    int p = t & (HWP - 1);
    int n = t >> 16;
    float* sp = src + (long)n * KKC * HWP + p;
    float mx = -INFINITY;
    for (int k = 0; k < KKC; ++k) mx = fmaxf(mx, sp[(long)k * HWP]);
    float s = 0.f;
    for (int k = 0; k < KKC; ++k) s += __expf(sp[(long)k * HWP] - mx);
    float add = log_belief[t] - (mx + __logf(s));
    for (int k = 0; k < KKC; ++k) sp[(long)k * HWP] += add;
  }
}

// ---------------------------------------------------------------------------
// Gather form of the scatter-logsumexp: for output pixel q, online-logsumexp
// over the 49 kernel taps of src[k, q - offset(k)].
// ---------------------------------------------------------------------------
__global__ void gather_kernel(const float* __restrict__ src, float* __restrict__ out) {
  int total = NB * HWP;
  for (int t = blockIdx.x * blockDim.x + threadIdx.x; t < total; t += gridDim.x * blockDim.x) {
    int p = t & (HWP - 1);
    int n = t >> 16;
    int oi = p >> 8, oj = p & 255;
    float m = -INFINITY, s = 0.f;
    for (int ky = 0; ky < 7; ++ky) {
      int si = oi - ky + 3;
      if (si < 0 || si >= HH) continue;
      for (int kx = 0; kx < 7; ++kx) {
        int sj = oj - kx + 3;
        if (sj < 0 || sj >= WW) continue;
        int k = ky * 7 + kx;
        float v = src[(long)(n * KKC + k) * HWP + si * WW + sj];
        if (v > m) { s = s * __expf(m - v) + 1.f; m = v; }
        else       { s += __expf(v - m); }
      }
    }
    out[t] = m + __logf(s);
  }
}

// ---------------------------------------------------------------------------
extern "C" void kernel_launch(void* const* d_in, const int* in_sizes, int n_in,
                              void* d_out, int out_size, void* d_ws, size_t ws_size,
                              hipStream_t stream) {
  (void)in_sizes; (void)n_in; (void)out_size; (void)ws_size;
  const float* log_belief = (const float*)d_in[0];
  const float* sem        = (const float*)d_in[1];
  const float* action     = (const float*)d_in[2];
  const float* w_in       = (const float*)d_in[3];
  const float* b_in       = (const float*)d_in[4];
  const float* w_d3       = (const float*)d_in[5];
  const float* b_d3       = (const float*)d_in[6];
  const float* w_d1       = (const float*)d_in[7];
  const float* b_d1       = (const float*)d_in[8];
  const float* w_out      = (const float*)d_in[9];
  const float* b_out      = (const float*)d_in[10];
  const float* w_k        = (const float*)d_in[11];
  const float* b_k        = (const float*)d_in[12];
  const float* w_a1       = (const float*)d_in[13];
  const float* b_a1       = (const float*)d_in[14];
  const float* w_a2       = (const float*)d_in[15];
  const float* b_a2       = (const float*)d_in[16];
  float* out = (float*)d_out;

  // Carve workspace (offsets in bytes, 256B aligned)
  char* ws = (char*)d_ws;
  size_t off = 0;
  auto carve = [&](size_t bytes) { void* p = ws + off; off += (bytes + 255) & ~(size_t)255; return p; };
  float*          x       = (float*)         carve((size_t)NB * HWP * DCH * 4);   // 64 MB NHWC
  unsigned short* h0      = (unsigned short*)carve((size_t)NB * HWP * HIDC * 2);  // 64 MB bf16
  unsigned short* h1      = (unsigned short*)carve((size_t)NB * HWP * HIDC * 2);  // 64 MB bf16
  float*          src     = (float*)         carve((size_t)NB * KKC * HWP * 4);   // 98 MB
  float*          lang    = (float*)         carve((size_t)NB * 64 * 4);
  unsigned short* blob_in = (unsigned short*)carve(1 * 4 * 512 * 2);
  unsigned short* blob_d3 = (unsigned short*)carve(18 * 4 * 512 * 2);
  unsigned short* blob_d1 = (unsigned short*)carve(18 * 4 * 512 * 2);
  unsigned short* blob_wo = (unsigned short*)carve(2 * 2 * 512 * 2);
  unsigned short* blob_wk = (unsigned short*)carve(1 * 4 * 512 * 2);
  unsigned short* h2 = h0;  // reuse: h0 consumed before h2 is produced

  // 1) weight repacks (tiny)
  repack_weights<<<8, 256, 0, stream>>>(w_in, blob_in, HIDC, DCH, 1, 1);
  repack_weights<<<64, 256, 0, stream>>>(w_d3, blob_d3, HIDC, HIDC, 3, 3);
  repack_weights<<<64, 256, 0, stream>>>(w_d1, blob_d1, HIDC, HIDC, 3, 3);
  repack_weights<<<8, 256, 0, stream>>>(w_out, blob_wo, DCH, HIDC, 1, 1);
  repack_weights<<<8, 256, 0, stream>>>(w_k, blob_wk, KKC, DCH, 1, 1);

  // 2) language head (tiny MLP + log_softmax)
  lang_kernel<<<NB, HIDC, 0, stream>>>(action, w_a1, b_a1, w_a2, b_a2, lang);

  // 3) 2x2 maxpool NCHW -> NHWC
  maxpool_kernel<<<8192, 256, 0, stream>>>(sem, x);

  // 4) 1x1 conv in (WMMA), 5)+6) dilated 3x3 convs (WMMA + TDM), 7) fused tail
  dim3 gconv(4096), bconv(256);  // 8 waves/block * 4096 blocks = 32768 M-tiles
  conv1x1_in_kernel<<<gconv, bconv, 0, stream>>>(x, blob_in, b_in, h0);
  conv3x3_kernel<<<gconv, bconv, 0, stream>>>(h0, blob_d3, b_d3, h1, 3);
  conv3x3_kernel<<<gconv, bconv, 0, stream>>>(h1, blob_d1, b_d1, h2, 1);
  out_motion_kernel<<<gconv, bconv, 0, stream>>>(h2, x, blob_wo, b_out, blob_wk, b_k, lang, src);

  // 8) per-pixel log-softmax fold, 9) 7x7 gather logsumexp
  normalize_kernel<<<2048, 256, 0, stream>>>(log_belief, src);
  gather_kernel<<<2048, 256, 0, stream>>>(src, out);
}
